// MultiheadAttention_36438502539830
// MI455X (gfx1250) — matrix operand order repaired and checked by reading, not compiled, not run on hardware
//
#include <hip/hip_runtime.h>
#include <hip/hip_bf16.h>
#include <cmath>

// Problem constants (from reference): B=2, T=2048, E=1024, P=1024, H=16, HD=64
#define B_  2
#define T_  2048
#define E_  1024
#define P_  1024
#define H_  16
#define HD_ 64

typedef __attribute__((ext_vector_type(16))) __bf16 v16bf;
typedef __attribute__((ext_vector_type(8)))  __bf16 v8bf;
typedef __attribute__((ext_vector_type(8)))  float  v8f;

typedef __attribute__((ext_vector_type(4))) unsigned int u32x4;
typedef __attribute__((ext_vector_type(8))) int          i32x8;
typedef __attribute__((ext_vector_type(4))) int          i32x4;

// ---------------------------------------------------------------------------
// WMMA fragment loaders (wave32; A and B share a contiguous k order)
//
// A (16x32 bf16): lane L (L<16) holds row M=L, K = 0..15;
//                 lane L+16 holds row M=L, K = 16..31.
// B (32x16 bf16): lane L holds column N=L&15, K = 16*(L>>4) + {0..15}.
// C/D (16x16 f32): VGPR r holds row M = 8*(lane>>4) + r, column N = lane&15.
// ---------------------------------------------------------------------------

__device__ inline v16bf load_a_bf16(const __bf16* row, int k0, int hi) {
  const __bf16* p = row + k0 + hi * 16;
  v8bf l = *(const v8bf*)(p);
  v8bf h = *(const v8bf*)(p + 8);
  v16bf a;
#pragma unroll
  for (int i = 0; i < 8; ++i) { a[i] = l[i]; a[8 + i] = h[i]; }
  return a;
}

__device__ inline v16bf load_a_f32(const float* row, int k0, int hi) {
  const float* p = row + k0 + hi * 16;
  v8f l = *(const v8f*)(p);
  v8f h = *(const v8f*)(p + 8);
  v16bf a;
#pragma unroll
  for (int i = 0; i < 8; ++i) { a[i] = (__bf16)l[i]; a[8 + i] = (__bf16)h[i]; }
  return a;
}

__device__ inline v16bf load_b_bf16(const __bf16* row, int k0, int hi) {
  return *(const v16bf*)(row + k0 + hi * 16);
}

__device__ inline v16bf load_b_f32(const float* row, int k0, int hi) {
  const float* p = row + k0 + hi * 16;
  v8f l = *(const v8f*)(p);
  v8f h = *(const v8f*)(p + 8);
  v16bf b;
#pragma unroll
  for (int i = 0; i < 8; ++i) { b[i] = (__bf16)l[i]; b[8 + i] = (__bf16)h[i]; }
  return b;
}

// ---------------------------------------------------------------------------
// Tensor Data Mover: 2D tile (rows x row_elems) global -> LDS.
// Builds the D# per ISA 8.3/8.4: group0 = {count=1, lds_addr, global_addr,
// type=2}; group1 = {data_size, tensor_dim0/1, tile_dim0/1, dim0_stride}.
// dsz_code: 0=1B 1=2B 2=4B 3=8B. Issued once per wave; tracked by TENSORcnt.
// ---------------------------------------------------------------------------
__device__ inline void tdm_load_2d_to_lds(const void* gptr, unsigned lds_off,
                                          unsigned row_elems, unsigned rows,
                                          unsigned stride_elems, unsigned dsz_code)
{
  unsigned long long ga = (unsigned long long)(uintptr_t)gptr;
  u32x4 g0;
  g0[0] = 1u;                                              // count=1, user desc
  g0[1] = lds_off;                                         // lds_addr (bytes)
  g0[2] = (unsigned)(ga & 0xffffffffull);                  // global_addr[31:0]
  g0[3] = (unsigned)((ga >> 32) & 0x01ffffffull) | (2u << 30); // [56:32], type=2
  i32x8 g1;
  g1[0] = (int)(dsz_code << 16);                           // wg_mask=0, data_size
  g1[1] = (int)((row_elems & 0xffffu) << 16);              // tensor_dim0[15:0]
  g1[2] = (int)(((row_elems >> 16) & 0xffffu) |            // tensor_dim0[31:16]
                ((rows & 0xffffu) << 16));                 // tensor_dim1[15:0]
  g1[3] = (int)(((rows >> 16) & 0xffffu) |                 // tensor_dim1[31:16]
                ((row_elems & 0xffffu) << 16));            // tile_dim0
  g1[4] = (int)(rows & 0xffffu);                           // tile_dim1, tile_dim2=0
  g1[5] = (int)stride_elems;                               // dim0_stride[31:0]
  g1[6] = 0;                                               // dim0_stride[47:32]
  g1[7] = 0;
  i32x4 z4 = {0, 0, 0, 0};
#if defined(__clang_major__) && (__clang_major__ >= 23)
  i32x8 z8 = {0, 0, 0, 0, 0, 0, 0, 0};
  __builtin_amdgcn_tensor_load_to_lds(g0, g1, z4, z4, z8, 0);
#else
  __builtin_amdgcn_tensor_load_to_lds(g0, g1, z4, z4, 0);
#endif
}

__device__ inline unsigned lds_offset_of(const void* p) {
  // Flat addresses in the LDS aperture use addr[31:0] as the LDS byte offset.
  return (unsigned)(uintptr_t)p;
}

// ---------------------------------------------------------------------------
// Kernel 1: input projection  proj = x @ W^T + bias  (M=B*T, N=P, K=E)
// A strip (16 x E fp32, 64KB) staged into LDS by the TDM; 8 waves share it.
//   layoutT==0 : qp/kp  [B,H,T,HD]
//   layoutT==1 : vpT    [B,H,HD,T]
// ---------------------------------------------------------------------------
__global__ void __launch_bounds__(256)
proj_kernel(const float* __restrict__ x, const float* __restrict__ W,
            const float* __restrict__ bias, __bf16* __restrict__ out, int layoutT)
{
  extern __shared__ __align__(128) float smemA[];   // 16 * E_ floats
  const int lane = threadIdx.x & 31;
  const int wid  = threadIdx.x >> 5;
  const int lo = lane & 15, hi = lane >> 4;
  const int tileM = blockIdx.x * 16;
  const int tileN = blockIdx.y * 128 + wid * 16;

  if (wid == 0) {
    tdm_load_2d_to_lds(x + (size_t)tileM * E_, lds_offset_of(smemA),
                       E_, 16, E_, 2u /*4B*/);
    __builtin_amdgcn_s_wait_tensorcnt(0);
  }
  __syncthreads();

  const float* arow = smemA + (size_t)lo * E_;
  const float* brow = W + (size_t)(tileN + lo) * E_;

  v8f acc = {};
  for (int k0 = 0; k0 < E_; k0 += 32) {
    __builtin_prefetch(brow + k0 + 64, 0, 0);
    v16bf a  = load_a_f32(arow, k0, hi);
    v16bf bm = load_b_f32(brow, k0, hi);
    acc = __builtin_amdgcn_wmma_f32_16x16x32_bf16(false, a, false, bm,
                                                  (short)0, acc, false, false);
  }

  const int n = tileN + lo;
  const int h = n >> 6, d = n & 63;
  const float bval = bias[n];
#pragma unroll
  for (int r = 0; r < 8; ++r) {
    const int m  = tileM + hi * 8 + r;
    const int bb = m / T_, t = m % T_;
    const float v = acc[r] + bval;
    const size_t idx = (layoutT == 0)
        ? ((((size_t)bb * H_ + h) * T_ + t) * HD_ + d)
        : ((((size_t)bb * H_ + h) * HD_ + d) * T_ + t);
    out[idx] = (__bf16)v;
  }
}

// ---------------------------------------------------------------------------
// Kernel 2: fused QK^T * 1/sqrt(HD) + causal/pad mask + row softmax.
// One block = one (b,h, 16-row q strip). Scores live in LDS (16 x 2064 fp32,
// padded stride so adjacent rows (one softmax half-wave each) hit disjoint banks).
// Causality: only tiles tk0 <= tq0+15 are computed (blockIdx.y+1 tiles);
// columns >= tq0+16 are written as exact 0.0f.
// ---------------------------------------------------------------------------
#define SSTRIDE 2064

__global__ void __launch_bounds__(256)
qk_softmax_kernel(const __bf16* __restrict__ qp, const __bf16* __restrict__ kp,
                  const float* __restrict__ attn_mask, float* __restrict__ att)
{
  extern __shared__ __align__(128) float srow[];    // 16 * SSTRIDE floats
  __shared__ float invrow[16];

  const int tid  = threadIdx.x;
  const int lane = tid & 31, wid = tid >> 5;
  const int lo = lane & 15, hi = lane >> 4;
  const int bh = blockIdx.x;
  const int b  = bh / H_;
  const int tq0 = blockIdx.y * 16;
  const int ntiles = blockIdx.y + 1;      // K tiles touching the causal region
  const int kvalid = tq0 + 16;

  const __bf16* qbase = qp + (size_t)bh * T_ * HD_;
  const __bf16* kbase = kp + (size_t)bh * T_ * HD_;
  const __bf16* arow  = qbase + (size_t)(tq0 + lo) * HD_;
  const v16bf a0 = load_a_bf16(arow, 0, hi);
  const v16bf a1 = load_a_bf16(arow, 32, hi);

  for (int tile = wid; tile < ntiles; tile += 8) {   // wave-uniform loop
    const int tk0 = tile * 16;
    const __bf16* brow = kbase + (size_t)(tk0 + lo) * HD_;
    v8f acc = {};
    acc = __builtin_amdgcn_wmma_f32_16x16x32_bf16(false, a0, false,
              load_b_bf16(brow, 0, hi),  (short)0, acc, false, false);
    acc = __builtin_amdgcn_wmma_f32_16x16x32_bf16(false, a1, false,
              load_b_bf16(brow, 32, hi), (short)0, acc, false, false);

    const int tk = tk0 + lo;
    const bool kvm = attn_mask[(size_t)b * T_ + tk] != 0.0f;
#pragma unroll
    for (int r = 0; r < 8; ++r) {
      const int tq = tq0 + hi * 8 + r;
      const float s = acc[r] * 0.125f;               // 1/sqrt(64)
      const bool keep = kvm && (tk <= tq);
      srow[(hi * 8 + r) * SSTRIDE + tk] = keep ? s : -INFINITY;
    }
  }
  __syncthreads();

  // Row softmax: 16 threads per row (a 16-lane half-wave group).
  {
    const int row = tid >> 4, sub = tid & 15;
    float* prow = srow + row * SSTRIDE;
    float m = -INFINITY;
    for (int c = sub; c < kvalid; c += 16) m = fmaxf(m, prow[c]);
#pragma unroll
    for (int off = 8; off > 0; off >>= 1) m = fmaxf(m, __shfl_xor(m, off, 32));
    float s = 0.0f;
    for (int c = sub; c < kvalid; c += 16) {
      const float e = __expf(prow[c] - m);
      prow[c] = e;
      s += e;
    }
#pragma unroll
    for (int off = 8; off > 0; off >>= 1) s += __shfl_xor(s, off, 32);
    if (sub == 0) invrow[row] = 1.0f / s;
  }
  __syncthreads();

  // Coalesced writeback of the 16 normalized rows (zeros above the diagonal).
  float* orow = att + ((size_t)bh * T_ + tq0) * T_;
  for (int idx = tid; idx < 16 * T_; idx += 256) {
    const int r = idx >> 11;            // T_ == 2048
    const int c = idx & (T_ - 1);
    const float v = (c < kvalid) ? srow[r * SSTRIDE + c] * invrow[r] : 0.0f;
    orow[(size_t)r * T_ + c] = v;
  }
}

// ---------------------------------------------------------------------------
// Kernel 3: y = att @ vp.  Per (b,h): M=T, N=HD=64, K=T. Causal truncation:
// att[tq, k>tq] == 0, so the K loop stops at the diagonal.
// Output y in [B,T,P] bf16 (fused head transpose).
// ---------------------------------------------------------------------------
__global__ void __launch_bounds__(256)
av_kernel(const float* __restrict__ att, const __bf16* __restrict__ vpT,
          __bf16* __restrict__ y)
{
  const int lane = threadIdx.x & 31;
  const int wid  = threadIdx.x >> 5;
  const int lo = lane & 15, hi = lane >> 4;
  const int bh = blockIdx.x;
  const int b  = bh >> 4, h = bh & 15;     // H_ == 16
  const int tq0 = blockIdx.y * 32 + (wid >> 2) * 16;
  const int d0  = (wid & 3) * 16;

  const float*  arow = att + (size_t)bh * T_ * T_ + (size_t)(tq0 + lo) * T_;
  const __bf16* brow = vpT + (size_t)bh * HD_ * T_ + (size_t)(d0 + lo) * T_;

  const int kend = ((tq0 + 16) + 31) & ~31;

  v8f acc = {};
  for (int k0 = 0; k0 < kend; k0 += 32) {
    v16bf a  = load_a_f32(arow, k0, hi);
    v16bf bm = load_b_bf16(brow, k0, hi);
    acc = __builtin_amdgcn_wmma_f32_16x16x32_bf16(false, a, false, bm,
                                                  (short)0, acc, false, false);
  }

#pragma unroll
  for (int r = 0; r < 8; ++r) {
    const int t = tq0 + hi * 8 + r;
    const int d = d0 + lo;
    y[((size_t)b * T_ + t) * P_ + h * HD_ + d] = (__bf16)acc[r];
  }
}

// ---------------------------------------------------------------------------
// Kernel 4: out = y @ Wp^T + bp  (M=B*T, N=E, K=P), fp32 result to d_out.
// A strip (16 x P bf16, 32KB) staged into LDS by the TDM.
// ---------------------------------------------------------------------------
__global__ void __launch_bounds__(256)
out_kernel(const __bf16* __restrict__ y, const float* __restrict__ Wp,
           const float* __restrict__ bp, float* __restrict__ out)
{
  extern __shared__ __align__(128) __bf16 smemY[];   // 16 * P_ bf16
  const int lane = threadIdx.x & 31;
  const int wid  = threadIdx.x >> 5;
  const int lo = lane & 15, hi = lane >> 4;
  const int tileM = blockIdx.x * 16;
  const int tileN = blockIdx.y * 128 + wid * 16;

  if (wid == 0) {
    tdm_load_2d_to_lds(y + (size_t)tileM * P_, lds_offset_of(smemY),
                       P_, 16, P_, 1u /*2B*/);
    __builtin_amdgcn_s_wait_tensorcnt(0);
  }
  __syncthreads();

  const __bf16* arow = smemY + (size_t)lo * P_;
  const float*  brow = Wp + (size_t)(tileN + lo) * P_;

  v8f acc = {};
  for (int k0 = 0; k0 < P_; k0 += 32) {
    __builtin_prefetch(brow + k0 + 64, 0, 0);
    v16bf a  = load_a_bf16(arow, k0, hi);
    v16bf bm = load_b_f32(brow, k0, hi);
    acc = __builtin_amdgcn_wmma_f32_16x16x32_bf16(false, a, false, bm,
                                                  (short)0, acc, false, false);
  }

  const int n = tileN + lo;
  const float bval = bp[n];
#pragma unroll
  for (int r = 0; r < 8; ++r) {
    const int m = tileM + hi * 8 + r;
    out[(size_t)m * E_ + n] = acc[r] + bval;
  }
}

// ---------------------------------------------------------------------------
// Host launcher
// ---------------------------------------------------------------------------
extern "C" void kernel_launch(void* const* d_in, const int* in_sizes, int n_in,
                              void* d_out, int out_size, void* d_ws, size_t ws_size,
                              hipStream_t stream)
{
  const float* q         = (const float*)d_in[0];
  const float* k         = (const float*)d_in[1];
  const float* v         = (const float*)d_in[2];
  const float* attn_mask = (const float*)d_in[3];
  const float* Wq        = (const float*)d_in[4];
  const float* bq        = (const float*)d_in[5];
  const float* Wk        = (const float*)d_in[6];
  const float* bk        = (const float*)d_in[7];
  const float* Wv        = (const float*)d_in[8];
  const float* bv        = (const float*)d_in[9];
  const float* Wp        = (const float*)d_in[10];
  const float* bp        = (const float*)d_in[11];

  float* out = (float*)d_out;                       // [B,T,E]
  float* att = out + (size_t)B_ * T_ * E_;          // [B,H,T,T]

  const size_t proj_elems = (size_t)B_ * H_ * T_ * HD_;   // 4M bf16 each
  __bf16* qp  = (__bf16*)d_ws;
  __bf16* kp  = qp  + proj_elems;
  __bf16* vpT = kp  + proj_elems;
  __bf16* y   = vpT + proj_elems;                   // [B,T,P] bf16

  const dim3 blk(256);

  const dim3 gproj(B_ * T_ / 16, P_ / 128);
  const size_t proj_lds = (size_t)16 * E_ * sizeof(float);       // 64 KB
  proj_kernel<<<gproj, blk, proj_lds, stream>>>(q, Wq, bq, qp, 0);
  proj_kernel<<<gproj, blk, proj_lds, stream>>>(k, Wk, bk, kp, 0);
  proj_kernel<<<gproj, blk, proj_lds, stream>>>(v, Wv, bv, vpT, 1);

  const size_t qk_lds = (size_t)16 * SSTRIDE * sizeof(float);    // 129 KB
  qk_softmax_kernel<<<dim3(B_ * H_, T_ / 16), blk, qk_lds, stream>>>(
      qp, kp, attn_mask, att);

  av_kernel<<<dim3(B_ * H_, T_ / 32), blk, 0, stream>>>(att, vpT, y);

  const size_t out_lds = (size_t)16 * P_ * sizeof(__bf16);       // 32 KB
  out_kernel<<<dim3(B_ * T_ / 16, E_ / 128), blk, out_lds, stream>>>(
      y, Wp, bp, out);
}
